// MultiViewBackbone_29532195127832
// MI455X (gfx1250) — compile-verified, matrix operand order checked by reading, not run on hardware
//
#include <hip/hip_runtime.h>
#include <stdint.h>

// ---------------------------------------------------------------------------
// Types
// ---------------------------------------------------------------------------
typedef unsigned short u16;
typedef unsigned int   u32;
typedef __attribute__((ext_vector_type(16))) __bf16          v16bf;
typedef __attribute__((ext_vector_type(8)))  float           v8f;
typedef __attribute__((ext_vector_type(8)))  unsigned short  ushort8;

struct U16x16 { ushort8 lo, hi; };   // 32 bytes, reinterpreted as v16bf

__device__ __forceinline__ float bf2f(u16 v) {
    u32 u = ((u32)v) << 16;
    return __builtin_bit_cast(float, u);
}
__device__ __forceinline__ u16 f2bf(float f) {
    u32 u = __builtin_bit_cast(u32, f);
    u32 r = (u + 0x7FFFu + ((u >> 16) & 1u)) >> 16;   // round-nearest-even
    return (u16)r;
}

// ---------------------------------------------------------------------------
// Scatter voxel features -> dense bf16 grid + occupancy mask
// ---------------------------------------------------------------------------
__global__ void scatter_kernel(const float* __restrict__ vf,
                               const int* __restrict__ coors,
                               u16* __restrict__ x0, u16* __restrict__ mask0,
                               int N, int H, int W, int C)
{
    int idx = blockIdx.x * blockDim.x + threadIdx.x;
    if (idx >= N * C) return;
    int n = idx / C, c = idx % C;
    int z = coors[n * 4 + 1], y = coors[n * 4 + 2], x = coors[n * 4 + 3];
    size_t pos = ((size_t)z * H + y) * W + x;
    x0[pos * C + c] = f2bf(vf[idx]);
    if (c == 0) mask0[pos] = 0x3F80;   // bf16 1.0
}

// ---------------------------------------------------------------------------
// Pack fp32 weights (kd,kh,kw,ci,co) into WMMA B-fragment order (bf16).
// Fragment layout per 32x16 K-chunk: lane = n + 16*((koff>>3)&1),
// elem j = (koff>=16 ? 8 : 0) + (koff&7).  Padding region pre-zeroed.
// ---------------------------------------------------------------------------
__global__ void pack_weights_kernel(const float* __restrict__ w,
                                    u16* __restrict__ dst,
                                    int taps, int Cin, int Co)
{
    int idx = blockIdx.x * blockDim.x + threadIdx.x;
    int total = taps * Cin * Co;
    if (idx >= total) return;
    int co = idx % Co; int t2 = idx / Co; int ci = t2 % Cin; int t = t2 / Cin;
    int nKc = (Cin + 31) >> 5, nCoT = Co >> 4;
    int kc = ci >> 5, koff = ci & 31;
    int n = co & 15, coT = co >> 4;
    int lane = n + 16 * ((koff >> 3) & 1);
    int j = ((koff >= 16) ? 8 : 0) + (koff & 7);
    size_t off = ((((size_t)t * nCoT + coT) * nKc + kc) * 32 + lane) * 16 + j;
    dst[off] = f2bf(w[idx]);
}

// ---------------------------------------------------------------------------
// Downsampled occupancy mask: out = (conv(mask, ones, stride, pad) > 0)
// ---------------------------------------------------------------------------
__global__ void down_mask_kernel(const u16* __restrict__ min_, u16* __restrict__ mout,
                                 int Din, int Hin, int Win, int Do, int Ho, int Wo,
                                 int kd, int kh, int kw, int sd, int sh, int sw,
                                 int pd, int ph, int pw)
{
    int idx = blockIdx.x * blockDim.x + threadIdx.x;
    int P = Do * Ho * Wo;
    if (idx >= P) return;
    int ow = idx % Wo; int t = idx / Wo; int oh = t % Ho; int od = t / Ho;
    u16 r = 0;
    for (int kz = 0; kz < kd; ++kz) {
        int iz = od * sd - pd + kz; if (iz < 0 || iz >= Din) continue;
        for (int ky = 0; ky < kh; ++ky) {
            int iy = oh * sh - ph + ky; if (iy < 0 || iy >= Hin) continue;
            for (int kx = 0; kx < kw; ++kx) {
                int ix = ow * sw - pw + kx; if (ix < 0 || ix >= Win) continue;
                if (min_[((size_t)iz * Hin + iy) * Win + ix] != 0) r = 0x3F80;
            }
        }
    }
    mout[idx] = r;
}

// ---------------------------------------------------------------------------
// Implicit-GEMM conv via V_WMMA_F32_16X16X32_BF16.
// One wave32 computes a 16-position x 16-outchannel tile; K runs over
// taps * ceil(Cin/32) chunks.  Epilogue fuses BN + residual + ReLU + mask.
// ---------------------------------------------------------------------------
__global__ __launch_bounds__(128)
void conv_wmma_kernel(const u16* __restrict__ in, const u16* __restrict__ wpack,
                      const float* __restrict__ g, const float* __restrict__ bb,
                      const u16* __restrict__ mask, const u16* __restrict__ res,
                      u16* __restrict__ out,
                      int Din, int Hin, int Win, int Cin,
                      int Do, int Ho, int Wo, int Co,
                      int kd, int kh, int kw, int sd, int sh, int sw,
                      int pd, int ph, int pw)
{
    const int lane   = threadIdx.x & 31;
    const int waveId = threadIdx.x >> 5;
    const int P      = Do * Ho * Wo;
    const int nTiles = (P + 15) >> 4;
    const int tile   = blockIdx.x * 4 + waveId;
    if (tile >= nTiles) return;                 // wave-uniform exit (EXEC stays full)
    const int coT  = blockIdx.y;
    const int co0  = coT * 16;
    const int nKc  = (Cin + 31) >> 5;
    const int nCoT = Co >> 4;
    const int taps = kd * kh * kw;

    // A-fragment row for this lane (ISA 16-bit A layout: both lane halves hold M=lane&15)
    const int m     = lane & 15;
    const int khalf = lane >> 4;
    const int p     = tile * 16 + m;
    const bool pvalid = (p < P);
    const int pp = pvalid ? p : 0;
    const int ow = pp % Wo; int t1 = pp / Wo; const int oh = t1 % Ho; const int od = t1 / Ho;

    v8f acc = {};
    const ushort8 z8 = {};

    for (int t = 0; t < taps; ++t) {
        int kx = t % kw; int t2 = t / kw; int ky = t2 % kh; int kz = t2 / kh;
        int iz = od * sd - pd + kz;
        int iy = oh * sh - ph + ky;
        int ix = ow * sw - pw + kx;
        bool v = pvalid && (iz >= 0) && (iz < Din) && (iy >= 0) && (iy < Hin)
                        && (ix >= 0) && (ix < Win);
        const u16* aptr = in + ((size_t)((iz * Hin + iy) * Win + ix)) * Cin;
        const u16* bptr = wpack + ((size_t)(t * nCoT + coT)) * nKc * 512;
        // prefetch next tap's packed weights (speculative, gfx1250 global_prefetch_b8)
        __builtin_prefetch(bptr + (size_t)nCoT * nKc * 512, 0, 0);

        for (int kc = 0; kc < nKc; ++kc) {
            const int c0 = kc * 32 + khalf * 8;
            U16x16 ar;
            ar.lo = (v && (c0 + 8  <= Cin)) ? *(const ushort8*)(aptr + c0)      : z8;
            ar.hi = (v && (c0 + 24 <= Cin)) ? *(const ushort8*)(aptr + c0 + 16) : z8;
            v16bf afrag = __builtin_bit_cast(v16bf, ar);

            const u16* bp = bptr + (size_t)kc * 512 + lane * 16;
            U16x16 br;
            br.lo = *(const ushort8*)(bp);
            br.hi = *(const ushort8*)(bp + 8);
            v16bf bfrag = __builtin_bit_cast(v16bf, br);

            acc = __builtin_amdgcn_wmma_f32_16x16x32_bf16(
                      false, afrag, false, bfrag, (short)0, acc, false, false);
        }
    }

    // Epilogue: C layout -> VGPR r holds M=r (lanes 0-15) / M=r+8 (lanes 16-31), N=lane&15
    const int n  = lane & 15;
    const int mh = lane >> 4;
    const float gs  = g[co0 + n] * rsqrtf(1.0f + 1e-5f);
    const float bsc = bb[co0 + n];
    for (int r = 0; r < 8; ++r) {
        int mm = r + mh * 8;
        int pr = tile * 16 + mm;
        if (pr >= P) continue;
        float val = acc[r] * gs + bsc;
        if (res)  val += bf2f(res[(size_t)pr * Co + co0 + n]);
        val = val > 0.0f ? val : 0.0f;
        if (mask) val *= bf2f(mask[pr]);
        out[(size_t)pr * Co + co0 + n] = f2bf(val);
    }
}

// ---------------------------------------------------------------------------
// Output finalization (transpose/reshape, bf16 -> f32)
// ---------------------------------------------------------------------------
__global__ void finalize_bev_kernel(const u16* __restrict__ bev, float* __restrict__ out)
{
    int idx = blockIdx.x * blockDim.x + threadIdx.x;
    if (idx >= 32768) return;
    int c = idx >> 8, rem = idx & 255;              // out[c,h,w] ; rem = h*16+w
    out[idx] = bf2f(bev[(size_t)rem * 128 + c]);    // bev buffer [h*16+w, 128]
}

__global__ void finalize_rv_kernel(const u16* __restrict__ r, float* __restrict__ out)
{
    int idx = blockIdx.x * blockDim.x + threadIdx.x;
    if (idx >= 3072) return;
    int d = idx % 3, h = (idx / 3) & 15, w = (idx / 48) & 1, c = idx / 96;
    // r buffer [D=3,H=16,W=2,C=32]; out flat = ((c*2+w)*16+h)*3+d
    out[32768 + idx] = bf2f(r[(size_t)(((d * 16 + h) * 2) + w) * 32 + c]);
}

// ---------------------------------------------------------------------------
// Host side
// ---------------------------------------------------------------------------
static void launch_conv(hipStream_t s,
                        const u16* in, const u16* wp, const float* g, const float* b,
                        const u16* mask, const u16* res, u16* out,
                        int Din, int Hin, int Win, int Cin,
                        int Do, int Ho, int Wo, int Co,
                        int kd, int kh, int kw, int sd, int sh, int sw,
                        int pd, int ph, int pw)
{
    int P = Do * Ho * Wo;
    int nTiles = (P + 15) / 16;
    dim3 grid((nTiles + 3) / 4, Co / 16);
    conv_wmma_kernel<<<grid, 128, 0, s>>>(in, wp, g, b, mask, res, out,
                                          Din, Hin, Win, Cin, Do, Ho, Wo, Co,
                                          kd, kh, kw, sd, sh, sw, pd, ph, pw);
}

static void launch_mask(hipStream_t s, const u16* mi, u16* mo,
                        int Din, int Hin, int Win, int Do, int Ho, int Wo,
                        int kd, int kh, int kw, int sd, int sh, int sw,
                        int pd, int ph, int pw)
{
    int P = Do * Ho * Wo;
    down_mask_kernel<<<(P + 255) / 256, 256, 0, s>>>(mi, mo, Din, Hin, Win, Do, Ho, Wo,
                                                     kd, kh, kw, sd, sh, sw, pd, ph, pw);
}

struct ConvP  { const float* b; const float* g; const float* w; };
struct BlockP { ConvP c1, c2; };
struct PackJob { const float* w; u16* dst; int taps, Cin, Co; };

extern "C" void kernel_launch(void* const* d_in, const int* in_sizes, int n_in,
                              void* d_out, int out_size, void* d_ws, size_t ws_size,
                              hipStream_t stream)
{
    (void)in_sizes; (void)n_in; (void)out_size; (void)ws_size;

    const float* VF    = (const float*)d_in[0];
    const int*   COORS = (const int*)d_in[1];

    // params pytree, keys flattened alphabetically: bev,d1,d2,d3,m0,mb1a,mb1b,
    // mb2a,mb2b,mb3a,mb3b,mb4a,mb4b,rb1,rb2,rv0,rv1,rv2; conv={b,g,w},
    // block={b1,b2,g1,g2,w1,w2}
    auto getConv = [&](int base) -> ConvP {
        return { (const float*)d_in[base], (const float*)d_in[base + 1],
                 (const float*)d_in[base + 2] };
    };
    auto getBlock = [&](int base) -> BlockP {
        ConvP c1 = { (const float*)d_in[base],     (const float*)d_in[base + 2],
                     (const float*)d_in[base + 4] };
        ConvP c2 = { (const float*)d_in[base + 1], (const float*)d_in[base + 3],
                     (const float*)d_in[base + 5] };
        return { c1, c2 };
    };
    ConvP  bevP = getConv(2);
    ConvP  d1P = getConv(5), d2P = getConv(8), d3P = getConv(11), m0P = getConv(14);
    BlockP mb1a = getBlock(17), mb1b = getBlock(23);
    BlockP mb2a = getBlock(29), mb2b = getBlock(35);
    BlockP mb3a = getBlock(41), mb3b = getBlock(47);
    BlockP mb4a = getBlock(53), mb4b = getBlock(59);
    BlockP rb1  = getBlock(65), rb2  = getBlock(71);
    ConvP  rv0P = getConv(77), rv1P = getConv(80), rv2P = getConv(83);

    // ---- workspace arena -------------------------------------------------
    char*  ws  = (char*)d_ws;
    size_t cur = 0;
    auto alloc = [&](size_t nU16) -> u16* {
        u16* p = (u16*)(ws + cur);
        cur += ((nU16 * 2 + 255) / 256) * 256;
        return p;
    };

    // stage dims
    const int D0 = 25,  H0 = 128, W0 = 128; const int P0 = D0 * H0 * W0; // 409600
    const int D1 = 13,  H1 = 64,  W1 = 64;  const int P1 = D1 * H1 * W1; // 53248
    const int D2 = 7,   H2 = 32,  W2 = 32;  const int P2 = D2 * H2 * W2; // 7168
    const int D3 = 3,   H3 = 16,  W3 = 16;  const int P3 = D3 * H3 * W3; // 768

    u16* x0    = alloc((size_t)P0 * 128);
    u16* mask0 = alloc(P0);
    u16* s1a = alloc((size_t)P0 * 16); u16* s1t = alloc((size_t)P0 * 16); u16* s1b = alloc((size_t)P0 * 16);
    u16* mask1 = alloc(P1);
    u16* s2a = alloc((size_t)P1 * 32); u16* s2t = alloc((size_t)P1 * 32); u16* s2b = alloc((size_t)P1 * 32);
    u16* mask2 = alloc(P2);
    u16* s3a = alloc((size_t)P2 * 64); u16* s3t = alloc((size_t)P2 * 64); u16* s3b = alloc((size_t)P2 * 64);
    u16* mask3 = alloc(P3);
    u16* s4a = alloc((size_t)P3 * 128); u16* s4t = alloc((size_t)P3 * 128); u16* s4b = alloc((size_t)P3 * 128);
    u16* maskb = alloc(256);  u16* bevb = alloc(256 * 128);
    u16* mr0 = alloc(384); u16* r0a = alloc(384 * 32); u16* r0t = alloc(384 * 32); u16* r0b = alloc(384 * 32);
    u16* mr1 = alloc(192); u16* r1a = alloc(192 * 32); u16* r1t = alloc(192 * 32); u16* r1b = alloc(192 * 32);
    u16* mr2 = alloc(96);  u16* r2  = alloc(96 * 32);

    // packed weights
    size_t packStart = cur;
    PackJob jobs[32]; int nj = 0;
    auto packW = [&](const ConvP& c, int taps, int Cin, int Co) -> u16* {
        u16* dst = alloc((size_t)taps * (Co / 16) * ((Cin + 31) / 32) * 512);
        jobs[nj++] = { c.w, dst, taps, Cin, Co };
        return dst;
    };
    u16* wp_m0   = packW(m0P, 27, 128, 16);
    u16* wp_1a1  = packW(mb1a.c1, 27, 16, 16);  u16* wp_1a2 = packW(mb1a.c2, 27, 16, 16);
    u16* wp_1b1  = packW(mb1b.c1, 27, 16, 16);  u16* wp_1b2 = packW(mb1b.c2, 27, 16, 16);
    u16* wp_d1   = packW(d1P, 27, 16, 32);
    u16* wp_2a1  = packW(mb2a.c1, 27, 32, 32);  u16* wp_2a2 = packW(mb2a.c2, 27, 32, 32);
    u16* wp_2b1  = packW(mb2b.c1, 27, 32, 32);  u16* wp_2b2 = packW(mb2b.c2, 27, 32, 32);
    u16* wp_d2   = packW(d2P, 27, 32, 64);
    u16* wp_3a1  = packW(mb3a.c1, 27, 64, 64);  u16* wp_3a2 = packW(mb3a.c2, 27, 64, 64);
    u16* wp_3b1  = packW(mb3b.c1, 27, 64, 64);  u16* wp_3b2 = packW(mb3b.c2, 27, 64, 64);
    u16* wp_d3   = packW(d3P, 27, 64, 128);
    u16* wp_4a1  = packW(mb4a.c1, 27, 128, 128); u16* wp_4a2 = packW(mb4a.c2, 27, 128, 128);
    u16* wp_4b1  = packW(mb4b.c1, 27, 128, 128); u16* wp_4b2 = packW(mb4b.c2, 27, 128, 128);
    u16* wp_bev  = packW(bevP, 3, 128, 128);
    u16* wp_rv0  = packW(rv0P, 3, 128, 32);
    u16* wp_rb11 = packW(rb1.c1, 27, 32, 32);   u16* wp_rb12 = packW(rb1.c2, 27, 32, 32);
    u16* wp_rv1  = packW(rv1P, 3, 32, 32);
    u16* wp_rb21 = packW(rb2.c1, 27, 32, 32);   u16* wp_rb22 = packW(rb2.c2, 27, 32, 32);
    u16* wp_rv2  = packW(rv2P, 3, 32, 32);
    size_t packBytes = cur - packStart;

    // ---- zero-init (x0+mask0 contiguous; pack arena for K-padding) -------
    hipMemsetAsync(x0, 0, (size_t)P0 * 128 * 2 + (size_t)P0 * 2, stream);
    hipMemsetAsync(ws + packStart, 0, packBytes, stream);

    // ---- weight packing --------------------------------------------------
    for (int j = 0; j < nj; ++j) {
        int total = jobs[j].taps * jobs[j].Cin * jobs[j].Co;
        pack_weights_kernel<<<(total + 255) / 256, 256, 0, stream>>>(
            jobs[j].w, jobs[j].dst, jobs[j].taps, jobs[j].Cin, jobs[j].Co);
    }

    // ---- scatter ---------------------------------------------------------
    {
        int total = 60000 * 128;
        scatter_kernel<<<(total + 255) / 256, 256, 0, stream>>>(
            VF, COORS, x0, mask0, 60000, H0, W0, 128);
    }

    auto submBlock = [&](const BlockP& bp, u16* X, u16* T, u16* Y, const u16* mask,
                         u16* w1, u16* w2, int D, int H, int W, int C) {
        launch_conv(stream, X, w1, bp.c1.g, bp.c1.b, mask, nullptr, T,
                    D, H, W, C, D, H, W, C, 3, 3, 3, 1, 1, 1, 1, 1, 1);
        launch_conv(stream, T, w2, bp.c2.g, bp.c2.b, mask, X, Y,
                    D, H, W, C, D, H, W, C, 3, 3, 3, 1, 1, 1, 1, 1, 1);
    };

    // ---- stage 1 (C=16) --------------------------------------------------
    launch_conv(stream, x0, wp_m0, m0P.g, m0P.b, mask0, nullptr, s1a,
                D0, H0, W0, 128, D0, H0, W0, 16, 3, 3, 3, 1, 1, 1, 1, 1, 1);
    submBlock(mb1a, s1a, s1t, s1b, mask0, wp_1a1, wp_1a2, D0, H0, W0, 16);
    submBlock(mb1b, s1b, s1t, s1a, mask0, wp_1b1, wp_1b2, D0, H0, W0, 16);

    // ---- down 1 -> stage 2 (C=32) ---------------------------------------
    launch_mask(stream, mask0, mask1, D0, H0, W0, D1, H1, W1, 3, 3, 3, 2, 2, 2, 1, 1, 1);
    launch_conv(stream, s1a, wp_d1, d1P.g, d1P.b, mask1, nullptr, s2a,
                D0, H0, W0, 16, D1, H1, W1, 32, 3, 3, 3, 2, 2, 2, 1, 1, 1);
    submBlock(mb2a, s2a, s2t, s2b, mask1, wp_2a1, wp_2a2, D1, H1, W1, 32);
    submBlock(mb2b, s2b, s2t, s2a, mask1, wp_2b1, wp_2b2, D1, H1, W1, 32);

    // ---- down 2 -> stage 3 (C=64) ---------------------------------------
    launch_mask(stream, mask1, mask2, D1, H1, W1, D2, H2, W2, 3, 3, 3, 2, 2, 2, 1, 1, 1);
    launch_conv(stream, s2a, wp_d2, d2P.g, d2P.b, mask2, nullptr, s3a,
                D1, H1, W1, 32, D2, H2, W2, 64, 3, 3, 3, 2, 2, 2, 1, 1, 1);
    submBlock(mb3a, s3a, s3t, s3b, mask2, wp_3a1, wp_3a2, D2, H2, W2, 64);
    submBlock(mb3b, s3b, s3t, s3a, mask2, wp_3b1, wp_3b2, D2, H2, W2, 64);

    // ---- down 3 -> stage 4 (C=128), pad D = 0 ---------------------------
    launch_mask(stream, mask2, mask3, D2, H2, W2, D3, H3, W3, 3, 3, 3, 2, 2, 2, 0, 1, 1);
    launch_conv(stream, s3a, wp_d3, d3P.g, d3P.b, mask3, nullptr, s4a,
                D2, H2, W2, 64, D3, H3, W3, 128, 3, 3, 3, 2, 2, 2, 0, 1, 1);
    submBlock(mb4a, s4a, s4t, s4b, mask3, wp_4a1, wp_4a2, D3, H3, W3, 128);
    submBlock(mb4b, s4b, s4t, s4a, mask3, wp_4b1, wp_4b2, D3, H3, W3, 128);

    // ---- BEV head: (3,1,1) stride (2,1,1) pad 0 -> (1,16,16,128) --------
    launch_mask(stream, mask3, maskb, D3, H3, W3, 1, 16, 16, 3, 1, 1, 2, 1, 1, 0, 0, 0);
    launch_conv(stream, s4a, wp_bev, bevP.g, bevP.b, maskb, nullptr, bevb,
                D3, H3, W3, 128, 1, 16, 16, 128, 3, 1, 1, 2, 1, 1, 0, 0, 0);
    finalize_bev_kernel<<<(32768 + 255) / 256, 256, 0, stream>>>(bevb, (float*)d_out);

    // ---- RV head: chain of (1,1,3) stride (1,1,2) pad (0,0,1) -----------
    launch_mask(stream, mask3, mr0, D3, H3, W3, 3, 16, 8, 1, 1, 3, 1, 1, 2, 0, 0, 1);
    launch_conv(stream, s4a, wp_rv0, rv0P.g, rv0P.b, mr0, nullptr, r0a,
                D3, H3, W3, 128, 3, 16, 8, 32, 1, 1, 3, 1, 1, 2, 0, 0, 1);
    submBlock(rb1, r0a, r0t, r0b, mr0, wp_rb11, wp_rb12, 3, 16, 8, 32);

    launch_mask(stream, mr0, mr1, 3, 16, 8, 3, 16, 4, 1, 1, 3, 1, 1, 2, 0, 0, 1);
    launch_conv(stream, r0b, wp_rv1, rv1P.g, rv1P.b, mr1, nullptr, r1a,
                3, 16, 8, 32, 3, 16, 4, 32, 1, 1, 3, 1, 1, 2, 0, 0, 1);
    submBlock(rb2, r1a, r1t, r1b, mr1, wp_rb21, wp_rb22, 3, 16, 4, 32);

    launch_mask(stream, mr1, mr2, 3, 16, 4, 3, 16, 2, 1, 1, 3, 1, 1, 2, 0, 0, 1);
    launch_conv(stream, r1b, wp_rv2, rv2P.g, rv2P.b, mr2, nullptr, r2,
                3, 16, 4, 32, 3, 16, 2, 32, 1, 1, 3, 1, 1, 2, 0, 0, 1);
    finalize_rv_kernel<<<(3072 + 255) / 256, 256, 0, stream>>>(r2, (float*)d_out);
}
